// DGEBlock_89781996356026
// MI455X (gfx1250) — matrix operand order repaired and checked by reading, not compiled, not run on hardware
//
#include <hip/hip_runtime.h>
#include <math.h>
#include <stdint.h>

// ---------------------------------------------------------------------------
// CDNA5 (gfx1250) transformer block: bf16 WMMA GEMMs + WMMA attention,
// async global->LDS staging for GEMM A-tiles.
// ---------------------------------------------------------------------------

typedef __bf16 bf16_t;
typedef __attribute__((ext_vector_type(16))) __bf16 v16bf;
typedef __attribute__((ext_vector_type(8)))  __bf16 v8bf;
typedef __attribute__((ext_vector_type(4)))  __bf16 v4bf;
typedef __attribute__((ext_vector_type(8)))  float  v8f;

#define WMMA_BF16(a, b, c) \
  __builtin_amdgcn_wmma_f32_16x16x32_bf16(false, (a), false, (b), (short)0, (c), false, false)

static __device__ __forceinline__ v16bf cat16(v8bf lo, v8bf hi) {
  return __builtin_shufflevector(lo, hi, 0,1,2,3,4,5,6,7,8,9,10,11,12,13,14,15);
}

// A-matrix fragment (16x32 bf16), row-major LDS tile, pitch mult of 8.
// element e<8  -> k = e + 8*half ; e>=8 -> k = 8 + e + 8*half  (ISA 7.12.2)
static __device__ __forceinline__ v16bf load_a_frag(const bf16_t* rowp, int half) {
  v8bf lo = *(const v8bf*)(rowp + 8 * half);
  v8bf hi = *(const v8bf*)(rowp + 16 + 8 * half);
  return cat16(lo, hi);
}

// B-matrix fragment (32x16 bf16): lane holds column N=lane%16, K contiguous.
// element e<8  -> k = e + 8*half ; e>=8 -> k = 8 + e + 8*half.
static __device__ __forceinline__ v16bf load_b_frag(const bf16_t* colrun, int half) {
  v8bf lo = *(const v8bf*)(colrun + 8 * half);
  v8bf hi = *(const v8bf*)(colrun + 16 + 8 * half);
  return cat16(lo, hi);
}

// Async global -> LDS copy of 16 bytes (GLOBAL_LOAD_ASYNC_TO_LDS_B128, GVS mode).
// lds_dst: generic pointer to __shared__ (low 32 bits = workgroup LDS offset).
// gbase:   wave-uniform 64-bit global base (SGPR pair); voff: per-lane byte offset.
static __device__ __forceinline__ void async_load_b128_to_lds(void* lds_dst,
                                                              const void* gbase,
                                                              unsigned voff) {
  unsigned lds_off = (unsigned)(uintptr_t)lds_dst;
  asm volatile("global_load_async_to_lds_b128 %0, %1, %2"
               :: "v"(lds_off), "v"(voff), "s"(gbase)
               : "memory");
}
static __device__ __forceinline__ void wait_asynccnt0() {
  asm volatile("s_wait_asynccnt 0" ::: "memory");
}

// ---------------------------------------------------------------------------
// f32 -> bf16 cast (vectorized by 4)
// ---------------------------------------------------------------------------
__global__ __launch_bounds__(256) void cast_f32_bf16(const float* __restrict__ in,
                                                     bf16_t* __restrict__ out, int n4) {
  int i = blockIdx.x * 256 + threadIdx.x;
  if (i < n4) {
    float4 v = *(const float4*)(in + (size_t)i * 4);
    v4bf o;
    o[0] = (bf16_t)v.x; o[1] = (bf16_t)v.y; o[2] = (bf16_t)v.z; o[3] = (bf16_t)v.w;
    *(v4bf*)(out + (size_t)i * 4) = o;
  }
}

// ---------------------------------------------------------------------------
// LayerNorm over D=1024, output bf16. One block per row, 256 threads.
// ---------------------------------------------------------------------------
__global__ __launch_bounds__(256) void layernorm_bf16(const float* __restrict__ X,
                                                      const float* __restrict__ g,
                                                      const float* __restrict__ b,
                                                      bf16_t* __restrict__ out) {
  const int row = blockIdx.x;
  const int t = threadIdx.x;
  const float* xr = X + (size_t)row * 1024;
  float4 v = *(const float4*)(xr + t * 4);
  __shared__ float s1[256], s2[256];
  s1[t] = v.x + v.y + v.z + v.w;
  s2[t] = v.x * v.x + v.y * v.y + v.z * v.z + v.w * v.w;
  __syncthreads();
  for (int o = 128; o > 0; o >>= 1) {
    if (t < o) { s1[t] += s1[t + o]; s2[t] += s2[t + o]; }
    __syncthreads();
  }
  float mean = s1[0] * (1.0f / 1024.0f);
  float var  = s2[0] * (1.0f / 1024.0f) - mean * mean;
  float rstd = rsqrtf(var + 1e-5f);
  float4 gv = *(const float4*)(g + t * 4);
  float4 bv = *(const float4*)(b + t * 4);
  v4bf o;
  o[0] = (bf16_t)((v.x - mean) * rstd * gv.x + bv.x);
  o[1] = (bf16_t)((v.y - mean) * rstd * gv.y + bv.y);
  o[2] = (bf16_t)((v.z - mean) * rstd * gv.z + bv.z);
  o[3] = (bf16_t)((v.w - mean) * rstd * gv.w + bv.w);
  *(v4bf*)(out + (size_t)row * 1024 + t * 4) = o;
}

// ---------------------------------------------------------------------------
// bf16 WMMA GEMM: C = A[MxK] * B[KxN] (+bias, +epilogue)
// Block tile 256x128, K-tile 64, 8 waves (4 in M x 2 in N), wave tile 64x64
// (4x4 fragments). A-tile staged via async global->LDS; B-tile transposed
// through VGPRs into LDS [n][k].
// MODE 0: out_bf16 = acc + bias
// MODE 1: out_f32  = acc + bias + resid
// MODE 2: out_bf16 = gelu(acc + bias)
// ---------------------------------------------------------------------------
constexpr int G_AP = 72;  // LDS pitch (bf16 elems); 144B rows, 16B-aligned
constexpr int G_BP = 72;

template <int MODE>
__global__ __launch_bounds__(256) void gemm_bf16(const bf16_t* __restrict__ A,
                                                 const bf16_t* __restrict__ Bw,
                                                 const float* __restrict__ bias,
                                                 const float* __restrict__ resid,
                                                 float* __restrict__ outF,
                                                 bf16_t* __restrict__ outB,
                                                 int M, int N, int K) {
  __shared__ alignas(16) bf16_t As[256 * G_AP];   // 36.9 KB
  __shared__ alignas(16) bf16_t Bst[128 * G_BP];  // 18.4 KB, transposed [n][k]
  const int t = threadIdx.x;
  const int lane = t & 31, wave = t >> 5;
  const int lm = lane & 15, half = lane >> 4;
  const int wm = wave >> 1;  // 0..3 (M dir, 64 rows)
  const int wn = wave & 1;   // 0..1 (N dir, 64 cols)
  const int m_blk = blockIdx.y * 256;
  const int n_blk = blockIdx.x * 128;

  const bf16_t* Ablk = A + (size_t)m_blk * K;  // wave-uniform base for async loads

  v8f acc[4][4];
  for (int i = 0; i < 4; ++i)
    for (int j = 0; j < 4; ++j) acc[i][j] = (v8f){0.f, 0.f, 0.f, 0.f, 0.f, 0.f, 0.f, 0.f};

  for (int k0 = 0; k0 < K; k0 += 64) {
    // A tile 256x64: async global->LDS, 8 x b128 per thread
    for (int i = 0; i < 8; ++i) {
      int vid = i * 256 + t;      // 0..2047
      int row = vid >> 3;         // 0..255
      int c8  = (vid & 7) * 8;    // 0..56
      async_load_b128_to_lds(&As[row * G_AP + c8], Ablk,
                             (unsigned)(((unsigned)row * (unsigned)K + k0 + c8) * 2u));
    }
    // B tile 64x128 -> transposed LDS [n][k]
    for (int i = 0; i < 8; ++i) {
      int vid = i * 256 + t;      // 0..2047
      int k   = vid >> 5;         // 0..63
      int n4  = (vid & 31) * 4;   // 0..124
      v4bf v = *(const v4bf*)(Bw + (size_t)(k0 + k) * N + n_blk + n4);
      for (int j = 0; j < 4; ++j) Bst[(n4 + j) * G_BP + k] = v[j];
    }
    // Prefetch next B tile (weights are L2-resident; 192MB L2).
    if (k0 + 64 < K)
      __builtin_prefetch(Bw + (size_t)(k0 + 64 + (t >> 2)) * N + n_blk + (t & 3) * 32, 0, 1);
    wait_asynccnt0();
    __syncthreads();
    for (int kk = 0; kk < 64; kk += 32) {
      v16bf af[4], bfv[4];
      for (int mt = 0; mt < 4; ++mt)
        af[mt] = load_a_frag(&As[(wm * 64 + mt * 16 + lm) * G_AP + kk], half);
      for (int nt = 0; nt < 4; ++nt)
        bfv[nt] = load_b_frag(&Bst[(wn * 64 + nt * 16 + lm) * G_BP + kk], half);
      for (int mt = 0; mt < 4; ++mt)
        for (int nt = 0; nt < 4; ++nt)
          acc[mt][nt] = WMMA_BF16(af[mt], bfv[nt], acc[mt][nt]);
    }
    __syncthreads();
  }

  for (int mt = 0; mt < 4; ++mt)
    for (int nt = 0; nt < 4; ++nt) {
      int col = n_blk + wn * 64 + nt * 16 + lm;
      float bv = bias[col];
      for (int r = 0; r < 8; ++r) {
        int row = m_blk + wm * 64 + mt * 16 + r + 8 * half;
        size_t off = (size_t)row * N + col;
        float v = acc[mt][nt][r] + bv;
        if (MODE == 0) {
          outB[off] = (bf16_t)v;
        } else if (MODE == 1) {
          outF[off] = v + resid[off];
        } else {
          outB[off] = (bf16_t)(0.5f * v * (1.0f + erff(v * 0.70710678118654752f)));
        }
      }
    }
}

// ---------------------------------------------------------------------------
// Attention: B=32, C=256, H=16, Dh=64. qkv is bf16 [B*C, 3072] (q|k|v).
// One block per (batch, 32-row query block), looping heads. Outputs:
//   ctx bf16 [B*C, 1024] (heads merged), Aout f32 [B, C, C] = mean_h(attn).
// ---------------------------------------------------------------------------
__global__ __launch_bounds__(256) void attention_kernel(const bf16_t* __restrict__ qkv,
                                                        bf16_t* __restrict__ ctx,
                                                        float* __restrict__ Aout) {
  constexpr int C = 256, QKVP = 3072, D = 1024, Dh = 64, H = 16, RB = 32;
  constexpr int QP = 72, KP = 72, VP = 40, SP = 260;
  constexpr float SCALE = 0.125f;  // 1/sqrt(64)

  __shared__ alignas(16) bf16_t Qs[RB * QP];     // 4.6 KB
  __shared__ alignas(16) bf16_t Ks[64 * KP];     // 9.2 KB
  __shared__ alignas(16) bf16_t Vt[Dh * VP];     // 5.1 KB (transposed [dh][key32])
  __shared__ float S[RB * SP];                   // 33.3 KB
  __shared__ float red[RB][8];                   // 1 KB

  const int bi = blockIdx.x >> 3;
  const int c0 = (blockIdx.x & 7) * RB;
  const int t = threadIdx.x;
  const int lane = t & 31, wave = t >> 5;
  const int lm = lane & 15, half = lane >> 4;

  const int arow = t >> 3;           // softmax/A duty: row 0..31
  const int acb  = (t & 7) * 32;     // 32-col chunk

  float regA[32];
  for (int i = 0; i < 32; ++i) regA[i] = 0.f;

  const size_t qbase = (size_t)(bi * C + c0);  // query token base
  const size_t kbase = (size_t)(bi * C);       // key/value token base

  for (int h = 0; h < H; ++h) {
    {  // load Q tile 32x64
      int row = t >> 3, c8 = (t & 7) * 8;
      v8bf v = *(const v8bf*)(qkv + (qbase + row) * QKVP + h * Dh + c8);
      *(v8bf*)(&Qs[row * QP + c8]) = v;
    }
    __syncthreads();

    // ---- S = Q K^T : 4 iters of 64 keys; wave -> one 16x16 tile ----
    const int s_rt = wave >> 2;  // 0..1
    const int s_ct = wave & 3;   // 0..3
    for (int kb = 0; kb < 4; ++kb) {
      for (int i = 0; i < 2; ++i) {  // K tile 64x64
        int vid = i * 256 + t;
        int kr = vid >> 3, c8 = (vid & 7) * 8;
        v8bf v = *(const v8bf*)(qkv + (kbase + kb * 64 + kr) * QKVP + D + h * Dh + c8);
        *(v8bf*)(&Ks[kr * KP + c8]) = v;
      }
      __syncthreads();
      v8f sa = (v8f){0.f, 0.f, 0.f, 0.f, 0.f, 0.f, 0.f, 0.f};
      for (int kk = 0; kk < 64; kk += 32) {
        v16bf aq = load_a_frag(&Qs[(s_rt * 16 + lm) * QP + kk], half);
        v16bf bk = load_b_frag(&Ks[(s_ct * 16 + lm) * KP + kk], half);
        sa = WMMA_BF16(aq, bk, sa);
      }
      for (int r = 0; r < 8; ++r)
        S[(s_rt * 16 + r + 8 * half) * SP + kb * 64 + s_ct * 16 + lm] = sa[r];
      __syncthreads();
    }

    // ---- softmax(scale*S) rows + A accumulation ----
    {
      float* srow = &S[arow * SP];
      float m = -3.4e38f;
      for (int i = 0; i < 32; ++i) m = fmaxf(m, srow[acb + i]);
      red[arow][t & 7] = m;
      __syncthreads();
      float rowm = red[arow][0];
      for (int i = 1; i < 8; ++i) rowm = fmaxf(rowm, red[arow][i]);
      __syncthreads();
      float ssum = 0.f;
      for (int i = 0; i < 32; ++i) {
        float e = __expf((srow[acb + i] - rowm) * SCALE);
        srow[acb + i] = e;
        ssum += e;
      }
      red[arow][t & 7] = ssum;
      __syncthreads();
      float tot = 0.f;
      for (int i = 0; i < 8; ++i) tot += red[arow][i];
      float rinv = 1.0f / tot;
      for (int i = 0; i < 32; ++i) {
        float p = srow[acb + i] * rinv;
        srow[acb + i] = p;
        regA[i] += p * (1.0f / 16.0f);
      }
      __syncthreads();
    }

    // ---- ctx = P V : wave -> one 16x16 out tile, K-loop 8x32 keys ----
    const int c_rt = wave >> 2;  // row tile
    const int c_ct = wave & 3;   // dh tile
    v8f cacc = (v8f){0.f, 0.f, 0.f, 0.f, 0.f, 0.f, 0.f, 0.f};
    for (int kb = 0; kb < 8; ++kb) {
      {  // V tile 32 keys x 64 dh -> transposed LDS [dh][key]
        int kr = t >> 3, c8 = (t & 7) * 8;
        v8bf v = *(const v8bf*)(qkv + (kbase + kb * 32 + kr) * QKVP + 2 * D + h * Dh + c8);
        for (int j = 0; j < 8; ++j) Vt[(c8 + j) * VP + kr] = v[j];
      }
      __syncthreads();
      const float* sp = &S[(c_rt * 16 + lm) * SP + kb * 32];
      v16bf ap;
      for (int i = 0; i < 8; ++i) {
        ap[i]     = (bf16_t)sp[8 * half + i];
        ap[i + 8] = (bf16_t)sp[16 + 8 * half + i];
      }
      v16bf bv = load_b_frag(&Vt[(c_ct * 16 + lm) * VP], half);
      cacc = WMMA_BF16(ap, bv, cacc);
      __syncthreads();
    }
    for (int r = 0; r < 8; ++r) {
      int row = c_rt * 16 + r + 8 * half;
      ctx[(qbase + row) * D + h * Dh + c_ct * 16 + lm] = (bf16_t)cacc[r];
    }
    __syncthreads();
  }

  float* ao = Aout + (size_t)(bi * C + c0 + arow) * C + acb;
  for (int i = 0; i < 32; ++i) ao[i] = regA[i];
}

// ---------------------------------------------------------------------------
// Launch
// ---------------------------------------------------------------------------
extern "C" void kernel_launch(void* const* d_in, const int* in_sizes, int n_in,
                              void* d_out, int out_size, void* d_ws, size_t ws_size,
                              hipStream_t stream) {
  (void)in_sizes; (void)n_in; (void)out_size; (void)ws_size;
  const float* X     = (const float*)d_in[0];
  const float* Wqkv  = (const float*)d_in[1];
  const float* bqkv  = (const float*)d_in[2];
  const float* Wproj = (const float*)d_in[3];
  const float* bproj = (const float*)d_in[4];
  const float* g1    = (const float*)d_in[5];
  const float* be1   = (const float*)d_in[6];
  const float* g2    = (const float*)d_in[7];
  const float* be2   = (const float*)d_in[8];
  const float* Wff1  = (const float*)d_in[9];
  const float* bff1  = (const float*)d_in[10];
  const float* Wff2  = (const float*)d_in[11];
  const float* bff2  = (const float*)d_in[12];

  constexpr int Bz = 32, C = 256, D = 1024, DFF = 4096;
  constexpr int M = Bz * C;  // 8192 tokens

  float* Xout = (float*)d_out;
  float* Aout = Xout + (size_t)M * D;

  char* ws = (char*)d_ws;
  size_t off = 0;
  auto alloc = [&](size_t bytes) -> void* {
    void* p = ws + off;
    off = (off + bytes + 255) & ~(size_t)255;
    return p;
  };
  bf16_t* Wqkv_b  = (bf16_t*)alloc((size_t)D * 3 * D * 2);
  bf16_t* Wproj_b = (bf16_t*)alloc((size_t)D * D * 2);
  bf16_t* Wff1_b  = (bf16_t*)alloc((size_t)D * DFF * 2);
  bf16_t* Wff2_b  = (bf16_t*)alloc((size_t)DFF * D * 2);
  bf16_t* Xn_b    = (bf16_t*)alloc((size_t)M * D * 2);
  bf16_t* qkv_b   = (bf16_t*)alloc((size_t)M * 3 * D * 2);
  bf16_t* ctx_b   = (bf16_t*)alloc((size_t)M * D * 2);
  float*  X1      = (float*)alloc((size_t)M * D * 4);
  bf16_t* X2n_b   = (bf16_t*)alloc((size_t)M * D * 2);
  bf16_t* h_b     = (bf16_t*)alloc((size_t)M * DFF * 2);

  auto cast = [&](const float* src, bf16_t* dst, int n) {
    int n4 = n / 4;
    cast_f32_bf16<<<(n4 + 255) / 256, 256, 0, stream>>>(src, dst, n4);
  };
  cast(Wqkv, Wqkv_b, D * 3 * D);
  cast(Wproj, Wproj_b, D * D);
  cast(Wff1, Wff1_b, D * DFF);
  cast(Wff2, Wff2_b, DFF * D);

  // LN1 -> bf16
  layernorm_bf16<<<M, 256, 0, stream>>>(X, g1, be1, Xn_b);
  // qkv = Xn @ Wqkv + b  (bf16 out)
  gemm_bf16<0><<<dim3(3 * D / 128, M / 256), 256, 0, stream>>>(
      Xn_b, Wqkv_b, bqkv, nullptr, nullptr, qkv_b, M, 3 * D, D);
  // attention -> ctx bf16, A output
  attention_kernel<<<Bz * (C / 32), 256, 0, stream>>>(qkv_b, ctx_b, Aout);
  // X1 = X + ctx @ Wproj + b
  gemm_bf16<1><<<dim3(D / 128, M / 256), 256, 0, stream>>>(
      ctx_b, Wproj_b, bproj, X, X1, nullptr, M, D, D);
  // LN2 -> bf16
  layernorm_bf16<<<M, 256, 0, stream>>>(X1, g2, be2, X2n_b);
  // h = gelu(X2n @ Wff1 + b) (bf16 out)
  gemm_bf16<2><<<dim3(DFF / 128, M / 256), 256, 0, stream>>>(
      X2n_b, Wff1_b, bff1, nullptr, nullptr, h_b, M, DFF, D);
  // X_out = X1 + h @ Wff2 + b
  gemm_bf16<1><<<dim3(D / 128, M / 256), 256, 0, stream>>>(
      h_b, Wff2_b, bff2, X1, Xout, nullptr, M, D, DFF);
}
